// Interface_attention_47072841564866
// MI455X (gfx1250) — matrix operand order repaired and checked by reading, not compiled, not run on hardware
//
#include <hip/hip_runtime.h>

// ---------------------------------------------------------------------------
// MI455X (gfx1250) implementation.
// All GEMMs run on v_wmma_f32_16x16x32_bf16 (f32 accumulate). Problem is
// ~61 GFLOP and ~250 MB of traffic -> balanced at ~10us on 23.3 TB/s HBM,
// so matrix-core bf16 is the right precision/throughput point.
// Workspace layout: bf16-transposed weights (~0.5 MB) + Q fp32 (40.96 MB)
// + out_pre fp32 (40.96 MB) = ~82.5 MB.
// ---------------------------------------------------------------------------

constexpr int cN1 = 20000;
constexpr int cNN = 16;
constexpr int cNI = 128;
constexpr int cND = 128;
constexpr int cNH = 4;

typedef __attribute__((ext_vector_type(16))) __bf16 v16bf;
typedef __attribute__((ext_vector_type(8)))  float  v8f;

static __device__ __forceinline__ __bf16 f2bf(float f) {
  unsigned int u = __builtin_bit_cast(unsigned int, f);
  unsigned int r = (u + 0x7FFFu + ((u >> 16) & 1u)) >> 16;  // RNE
  unsigned short s = (unsigned short)r;
  return __builtin_bit_cast(__bf16, s);
}

static __device__ __forceinline__ float eluf(float x) {
  return x > 0.f ? x : (__expf(x) - 1.f);
}

// A-tile (16x32 bf16 activations) per ISA 16-bit A layout:
// lane holds row (lane&15); VGPR0..3 = K[half*8 .. +7], VGPR4..7 = K[16+half*8 .. +7]
static __device__ __forceinline__ v16bf load_a(const __bf16* lds, int stride,
                                               int kb, int lane) {
  const int row  = lane & 15;
  const int half = lane >> 4;
  const __bf16* p0 = lds + row * stride + kb + half * 8;
  const __bf16* p1 = p0 + 16;
  v16bf a;
#pragma unroll
  for (int i = 0; i < 8; ++i) { a[i] = p0[i]; a[8 + i] = p1[i]; }
  return a;
}

// B-tile (32x16 bf16 weights) from transposed weight Wt[out][inPad]:
// lane = column (lane&15); holds K[half*8 .. +7] then K[16+half*8 .. +7].
static __device__ __forceinline__ v16bf load_b(const __bf16* wt, int inPad,
                                               int nb, int kb, int lane) {
  const int col  = nb + (lane & 15);
  const int half = lane >> 4;
  const __bf16* p = wt + (size_t)col * inPad + kb + half * 8;
  v16bf b;
#pragma unroll
  for (int i = 0; i < 8; ++i) { b[i] = p[i]; b[8 + i] = p[16 + i]; }
  return b;
}

static __device__ __forceinline__ v8f gemm_tile(const __bf16* aLds, int aStride,
                                                const __bf16* wt, int inPad,
                                                int nb, int lane, int ksteps) {
  v8f c = {};
  for (int kt = 0; kt < ksteps; ++kt) {
    v16bf a = load_a(aLds, aStride, kt * 32, lane);
    v16bf b = load_b(wt, inPad, nb, kt * 32, lane);
    c = __builtin_amdgcn_wmma_f32_16x16x32_bf16(false, a, false, b,
                                                (short)0, c, false, false);
  }
  return c;
}

// --------------------------- weight prep -----------------------------------
__global__ void k_wconv(const float* __restrict__ W, __bf16* __restrict__ Wt,
                        int IN, int OUT, int INpad) {
  int idx = blockIdx.x * blockDim.x + threadIdx.x;
  if (idx >= OUT * INpad) return;
  int o = idx / INpad, i = idx - o * INpad;
  float v = (i < IN) ? W[(size_t)i * OUT + o] : 0.f;
  Wt[idx] = f2bf(v);
}

// --------------------------- Q MLP ------------------------------------------
__global__ __launch_bounds__(256) void k_query(
    const float* __restrict__ f1,
    const __bf16* __restrict__ wq1t, const float* __restrict__ qb1,
    const __bf16* __restrict__ wq2t, const float* __restrict__ qb2,
    float* __restrict__ Qws) {
  __shared__ __bf16 sA[16][128];
  __shared__ __bf16 sH[16][128];
  const int n0   = blockIdx.x * 16;
  const int tid  = threadIdx.x;
  const int lane = tid & 31;
  const int wave = tid >> 5;

#pragma unroll
  for (int i = 0; i < 8; ++i) {
    int e = tid * 8 + i;
    int r = e >> 7, c = e & 127;
    sA[r][c] = f2bf(f1[(size_t)(n0 + r) * cNI + c]);
  }
  __syncthreads();

  const int col = wave * 16 + (lane & 15);
  const int mh  = (lane >> 4) * 8;
  {  // H1 = elu(f1 @ q_w1 + q_b1)
    v8f c = gemm_tile(&sA[0][0], 128, wq1t, 128, wave * 16, lane, 4);
#pragma unroll
    for (int v = 0; v < 8; ++v)
      sH[mh + v][col] = f2bf(eluf(c[v] + qb1[col]));
  }
  __syncthreads();

#pragma unroll
  for (int t = 0; t < 4; ++t) {  // Q = H1 @ q_w2 + q_b2   (512 cols)
    int nb = t * 128 + wave * 16;
    v8f c  = gemm_tile(&sH[0][0], 128, wq2t, 128, nb, lane, 4);
    int cc = nb + (lane & 15);
#pragma unroll
    for (int v = 0; v < 8; ++v)
      Qws[(size_t)(n0 + mh + v) * (cNH * cND) + cc] = c[v] + qb2[cc];
  }
}

// --------------------------- neighbor pipeline ------------------------------
__global__ __launch_bounds__(256) void k_neighbors(
    const float* __restrict__ f2,  const float* __restrict__ x1,
    const float* __restrict__ x2,  const float* __restrict__ nuv1,
    const float* __restrict__ nuv2, const int* __restrict__ topk,
    const float* __restrict__ Qws,
    const __bf16* __restrict__ wg1t, const float* __restrict__ gb1,
    const __bf16* __restrict__ wg2t, const float* __restrict__ gb2,
    const __bf16* __restrict__ wk1t, const __bf16* __restrict__ wk2t,
    const __bf16* __restrict__ wv1t, const __bf16* __restrict__ wv2t,
    float* __restrict__ outPre) {
  __shared__ __bf16 sRL[16][32];
  __shared__ __bf16 sAct[16][128];
  __shared__ __bf16 sB[16][128];
  __shared__ float  sF2[16][128];
  __shared__ float  sK[16][128];
  __shared__ float  sV[16][128];
  __shared__ float  sMq[4][16];
  __shared__ int    sJ[16];
  __shared__ float  sGauss[16];

  const int n    = blockIdx.x;
  const int tid  = threadIdx.x;
  const int lane = tid & 31;
  const int wave = tid >> 5;

  if (tid < 16) sJ[tid] = topk[(size_t)n * cNN + tid];
  __syncthreads();

  // gather neighbor feature rows (L2-resident: features2 is only 10 MB)
#pragma unroll
  for (int i = 0; i < 8; ++i) {
    int e = tid * 8 + i;
    int k = e >> 7, d = e & 127;
    sF2[k][d] = f2[(size_t)sJ[k] * cND + d];
  }

  // geometry: rotate into local frame, build RL[16][12] (zero-pad K to 32)
  if (tid < 16) {
    const int k = tid;
    const int j = sJ[k];
    float M[9], o[3], rl[12];
#pragma unroll
    for (int i = 0; i < 9; ++i) M[i] = nuv1[(size_t)n * 9 + i];
#pragma unroll
    for (int i = 0; i < 3; ++i)
      o[i] = x2[(size_t)j * 3 + i] - x1[(size_t)n * 3 + i];
#pragma unroll
    for (int i = 0; i < 3; ++i)
      rl[i] = M[i * 3 + 0] * o[0] + M[i * 3 + 1] * o[1] + M[i * 3 + 2] * o[2];
#pragma unroll
    for (int r = 0; r < 3; ++r) {
      float w0 = nuv2[(size_t)j * 9 + r * 3 + 0];
      float w1 = nuv2[(size_t)j * 9 + r * 3 + 1];
      float w2 = nuv2[(size_t)j * 9 + r * 3 + 2];
#pragma unroll
      for (int i = 0; i < 3; ++i)
        rl[3 + r * 3 + i] = M[i * 3 + 0] * w0 + M[i * 3 + 1] * w1 + M[i * 3 + 2] * w2;
    }
    float d2 = o[0] * o[0] + o[1] * o[1] + o[2] * o[2];
    sGauss[k] = __expf(-d2 * 0.5f);  // RADIUS = 1
#pragma unroll
    for (int c = 0; c < 12; ++c) sRL[k][c] = f2bf(rl[c]);
#pragma unroll
    for (int c = 12; c < 32; ++c) sRL[k][c] = f2bf(0.f);
  }
  __syncthreads();

  const int col = wave * 16 + (lane & 15);
  const int mh  = (lane >> 4) * 8;

  {  // geo1 = elu(RL @ g_w1 + g_b1)   K=32 (padded from 12)
    v8f c = gemm_tile(&sRL[0][0], 32, wg1t, 32, wave * 16, lane, 1);
#pragma unroll
    for (int v = 0; v < 8; ++v)
      sAct[mh + v][col] = f2bf(eluf(c[v] + gb1[col]));
  }
  __syncthreads();
  {  // geo2 = (geo1 @ g_w2 + g_b2) * features_nn * gauss
    v8f c = gemm_tile(&sAct[0][0], 128, wg2t, 128, wave * 16, lane, 4);
#pragma unroll
    for (int v = 0; v < 8; ++v) {
      int k = mh + v;
      sB[k][col] = f2bf((c[v] + gb2[col]) * sF2[k][col] * sGauss[k]);
    }
  }
  __syncthreads();
  {  // K1 = elu(geo @ k_w1)
    v8f c = gemm_tile(&sB[0][0], 128, wk1t, 128, wave * 16, lane, 4);
#pragma unroll
    for (int v = 0; v < 8; ++v) sAct[mh + v][col] = f2bf(eluf(c[v]));
  }
  __syncthreads();
  {  // K = K1 @ k_w2
    v8f c = gemm_tile(&sAct[0][0], 128, wk2t, 128, wave * 16, lane, 4);
#pragma unroll
    for (int v = 0; v < 8; ++v) sK[mh + v][col] = c[v];
  }
  __syncthreads();
  {  // V1 = elu(geo @ v_w1)
    v8f c = gemm_tile(&sB[0][0], 128, wv1t, 128, wave * 16, lane, 4);
#pragma unroll
    for (int v = 0; v < 8; ++v) sAct[mh + v][col] = f2bf(eluf(c[v]));
  }
  __syncthreads();
  {  // V = V1 @ v_w2
    v8f c = gemm_tile(&sAct[0][0], 128, wv2t, 128, wave * 16, lane, 4);
#pragma unroll
    for (int v = 0; v < 8; ++v) sV[mh + v][col] = c[v];
  }
  __syncthreads();

  // Mq[h][k] = Q[h] . K[k], masked where topk == 0  (tiny -> VALU)
  if (tid < 64) {
    int h = tid >> 4, k = tid & 15;
    const float* q = Qws + (size_t)n * (cNH * cND) + (size_t)h * cND;
    float acc = 0.f;
#pragma unroll 4
    for (int d = 0; d < cND; ++d) acc += q[d] * sK[k][d];
    sMq[h][k] = (sJ[k] == 0) ? 0.f : acc;
  }
  __syncthreads();

  // out_pre[h*128+d] = sum_k Mq[h][k] * V[k][d]
#pragma unroll
  for (int rep = 0; rep < 2; ++rep) {
    int idx = tid + rep * 256;
    int h = idx >> 7, d = idx & 127;
    float acc = 0.f;
#pragma unroll
    for (int k = 0; k < cNN; ++k) acc += sMq[h][k] * sV[k][d];
    outPre[(size_t)n * (cNH * cND) + idx] = acc;
  }
}

// --------------------------- output MLP -------------------------------------
__global__ __launch_bounds__(256) void k_final(
    const float* __restrict__ outPre,
    const __bf16* __restrict__ wd1t, const __bf16* __restrict__ wd2t,
    float* __restrict__ out) {
  __shared__ __bf16 sA[16][512];
  __shared__ __bf16 sT[16][128];
  const int n0   = blockIdx.x * 16;
  const int tid  = threadIdx.x;
  const int lane = tid & 31;
  const int wave = tid >> 5;

#pragma unroll
  for (int i = 0; i < 32; ++i) {
    int e = tid * 32 + i;
    int r = e >> 9, c = e & 511;
    sA[r][c] = f2bf(outPre[(size_t)(n0 + r) * 512 + c]);
  }
  __syncthreads();

  const int col = wave * 16 + (lane & 15);
  const int mh  = (lane >> 4) * 8;
  {  // t = elu(out_pre @ d_w1)   K=512
    v8f c = gemm_tile(&sA[0][0], 512, wd1t, 512, wave * 16, lane, 16);
#pragma unroll
    for (int v = 0; v < 8; ++v) sT[mh + v][col] = f2bf(eluf(c[v]));
  }
  __syncthreads();
  {  // out = elu(t @ d_w2)
    v8f c = gemm_tile(&sT[0][0], 128, wd2t, 128, wave * 16, lane, 4);
#pragma unroll
    for (int v = 0; v < 8; ++v)
      out[(size_t)(n0 + mh + v) * cND + col] = eluf(c[v]);
  }
}

// --------------------------- launcher ---------------------------------------
extern "C" void kernel_launch(void* const* d_in, const int* in_sizes, int n_in,
                              void* d_out, int out_size, void* d_ws, size_t ws_size,
                              hipStream_t stream) {
  (void)in_sizes; (void)n_in; (void)out_size; (void)ws_size;
  const float* f1   = (const float*)d_in[0];
  const float* f2   = (const float*)d_in[1];
  const float* x1   = (const float*)d_in[2];
  const float* x2   = (const float*)d_in[3];
  const float* nv1  = (const float*)d_in[4];
  const float* nv2  = (const float*)d_in[5];
  const int*   topk = (const int*)d_in[6];
  const float* q_w1 = (const float*)d_in[7];
  const float* q_b1 = (const float*)d_in[8];
  const float* q_w2 = (const float*)d_in[9];
  const float* q_b2 = (const float*)d_in[10];
  const float* g_w1 = (const float*)d_in[11];
  const float* g_b1 = (const float*)d_in[12];
  const float* g_w2 = (const float*)d_in[13];
  const float* g_b2 = (const float*)d_in[14];
  const float* k_w1 = (const float*)d_in[15];
  const float* k_w2 = (const float*)d_in[16];
  const float* v_w1 = (const float*)d_in[17];
  const float* v_w2 = (const float*)d_in[18];
  const float* d_w1 = (const float*)d_in[19];
  const float* d_w2 = (const float*)d_in[20];

  char* ws = (char*)d_ws;
  auto take = [&](size_t bytes) -> char* {
    char* p = ws;
    ws += (bytes + 255) & ~(size_t)255;
    return p;
  };
  __bf16* wq1t = (__bf16*)take((size_t)128 * 128 * 2);
  __bf16* wq2t = (__bf16*)take((size_t)512 * 128 * 2);
  __bf16* wg1t = (__bf16*)take((size_t)128 * 32 * 2);
  __bf16* wg2t = (__bf16*)take((size_t)128 * 128 * 2);
  __bf16* wk1t = (__bf16*)take((size_t)128 * 128 * 2);
  __bf16* wk2t = (__bf16*)take((size_t)128 * 128 * 2);
  __bf16* wv1t = (__bf16*)take((size_t)128 * 128 * 2);
  __bf16* wv2t = (__bf16*)take((size_t)128 * 128 * 2);
  __bf16* wd1t = (__bf16*)take((size_t)128 * 512 * 2);
  __bf16* wd2t = (__bf16*)take((size_t)128 * 128 * 2);
  float*  Qws    = (float*)take((size_t)cN1 * 512 * 4);
  float*  outPre = (float*)take((size_t)cN1 * 512 * 4);

  auto conv = [&](const float* W, __bf16* Wt, int IN, int OUT, int INpad) {
    int total = OUT * INpad;
    k_wconv<<<(total + 255) / 256, 256, 0, stream>>>(W, Wt, IN, OUT, INpad);
  };
  conv(q_w1, wq1t, 128, 128, 128);
  conv(q_w2, wq2t, 128, 512, 128);
  conv(g_w1, wg1t, 12, 128, 32);
  conv(g_w2, wg2t, 128, 128, 128);
  conv(k_w1, wk1t, 128, 128, 128);
  conv(k_w2, wk2t, 128, 128, 128);
  conv(v_w1, wv1t, 128, 128, 128);
  conv(v_w2, wv2t, 128, 128, 128);
  conv(d_w1, wd1t, 512, 128, 512);
  conv(d_w2, wd2t, 128, 128, 128);

  k_query<<<cN1 / 16, 256, 0, stream>>>(f1, wq1t, q_b1, wq2t, q_b2, Qws);
  k_neighbors<<<cN1, 256, 0, stream>>>(f2, x1, x2, nv1, nv2, topk, Qws,
                                       wg1t, g_b1, wg2t, g_b2,
                                       wk1t, wk2t, wv1t, wv2t, outPre);
  k_final<<<cN1 / 16, 256, 0, stream>>>(outPre, wd1t, wd2t, (float*)d_out);
}
